// Self_Attention_11776800325784
// MI455X (gfx1250) — compile-verified, matrix operand order checked by reading, not run on hardware
//
#include <hip/hip_runtime.h>

// ---------------- problem constants ----------------
#define BATCH   4
#define SEQ     2048
#define DMODEL  1024
#define NH      16
#define DH      64
#define MTOT    (BATCH*SEQ)        // 8192 rows of x flattened

typedef __attribute__((ext_vector_type(16))) __bf16 v16bf;
typedef __attribute__((ext_vector_type(8)))  float  v8f;
typedef __attribute__((ext_vector_type(4)))  int    v4i_t;
typedef unsigned short u16;
typedef unsigned int   u32;

#define ASG __attribute__((address_space(1)))   // global
#define ASL __attribute__((address_space(3)))   // LDS

// ---------- CDNA5 feature detection (compile-safe guards) ----------
#if defined(__gfx1250__) && __has_builtin(__builtin_amdgcn_global_load_async_to_lds_b128)
#define HAVE_ASYNC 1
#else
#define HAVE_ASYNC 0
#endif

#if defined(__gfx1250__) && __has_builtin(__builtin_amdgcn_tensor_load_to_lds) && \
    __has_builtin(__builtin_amdgcn_s_wait_tensorcnt)
#define HAVE_TDM 1
#else
#define HAVE_TDM 0
#endif

// 16B global->LDS copy: async (no VGPR roundtrip, ASYNCcnt) when available.
// Builtin signature (probe-derived): (v4i global*, v4i lds*, imm offset, imm cpol).
__device__ __forceinline__ void async_cp16(const u16* g, u16* l) {
#if HAVE_ASYNC
  __builtin_amdgcn_global_load_async_to_lds_b128(
      (ASG v4i_t*)g, (ASL v4i_t*)l, 0, 0);
#else
  *(uint4*)l = *(const uint4*)g;
#endif
}

__device__ __forceinline__ void wait_async0() {
#if HAVE_ASYNC
#if __has_builtin(__builtin_amdgcn_s_wait_asynccnt)
  __builtin_amdgcn_s_wait_asynccnt(0);
#else
  asm volatile("s_wait_asynccnt 0x0" ::: "memory");
#endif
#endif
}

__device__ __forceinline__ u32 lds_byte_off(const void* p) {
  return (u32)(size_t)(ASL const void*)p;   // ptrtoint of AS(3) = LDS offset
}

#if HAVE_TDM
typedef __attribute__((ext_vector_type(4))) unsigned int tdm_v4u;
typedef __attribute__((ext_vector_type(8))) int          tdm_v8i;
typedef __attribute__((ext_vector_type(4))) int          tdm_v4i;

// One TDM descriptor: 2-D tile of 16-bit elements.
// tile_d0 elems/row, tile_d1 rows, row stride = stride0 elems (D# section 8).
__device__ __forceinline__ void tdm_load_2d(u32 lds_off, const void* gptr,
                                            u32 tile_d0, u32 tile_d1,
                                            u32 tens_d0, u32 tens_d1,
                                            u32 stride0) {
  unsigned long long ga = (unsigned long long)(size_t)gptr;
  tdm_v4u g0;
  g0[0] = 1u;                                      // count=1, user mode
  g0[1] = lds_off;                                 // lds_addr
  g0[2] = (u32)ga;                                 // global_addr[31:0]
  g0[3] = (u32)((ga >> 32) & 0x1FFFFFFu) | (2u << 30); // addr[56:32] | type=2
  tdm_v8i g1;
  g1[0] = (int)(1u << 16);                         // data_size=1 (2 bytes)
  g1[1] = (int)((tens_d0 & 0xFFFFu) << 16);        // tensor_dim0[15:0]
  g1[2] = (int)(((tens_d0 >> 16) & 0xFFFFu) | ((tens_d1 & 0xFFFFu) << 16));
  g1[3] = (int)(((tens_d1 >> 16) & 0xFFFFu) | ((tile_d0 & 0xFFFFu) << 16));
  g1[4] = (int)(tile_d1 & 0xFFFFu);                // tile_dim1 | tile_dim2=0
  g1[5] = (int)stride0;                            // tensor_dim0_stride[31:0]
  g1[6] = 0;
  g1[7] = 0;
  tdm_v4i z4 = {0, 0, 0, 0};
#if __clang_major__ >= 23
  tdm_v8i z8 = {0, 0, 0, 0, 0, 0, 0, 0};
  __builtin_amdgcn_tensor_load_to_lds(g0, g1, z4, z4, z8, 0);
#else
  __builtin_amdgcn_tensor_load_to_lds(g0, g1, z4, z4, 0);
#endif
}
__device__ __forceinline__ void wait_tensor0() {
  __builtin_amdgcn_s_wait_tensorcnt(0);
}
#endif

// ---------------- helpers ----------------
__device__ __forceinline__ u16 f2bf(float f) {
  u32 u = __float_as_uint(f);
  u32 r = u + 0x7FFFu + ((u >> 16) & 1u);   // round-to-nearest-even
  return (u16)(r >> 16);
}

// 16x32 bf16 WMMA A/B operand from row-major LDS tile (ISA 7.12.2 layout).
__device__ __forceinline__ v16bf load_op(const u16* base, int stride) {
  int lane = threadIdx.x & 31;
  int row  = lane & 15;
  int kb   = (lane >> 4) << 3;
  union { v16bf v; uint4 q[2]; } u;
  u.q[0] = *(const uint4*)(base + row * stride + kb);
  u.q[1] = *(const uint4*)(base + row * stride + 16 + kb);
  return u.v;
}

__device__ __forceinline__ v8f wmma_bf16(v16bf a, v16bf b, v8f c) {
  return __builtin_amdgcn_wmma_f32_16x16x32_bf16(
      false, a, false, b, (short)0, c, false, false);
}

__device__ __forceinline__ v8f v8f_zero() {
  v8f z = {0.f,0.f,0.f,0.f,0.f,0.f,0.f,0.f};
  return z;
}

// ---------------- kernel 0a: x fp32 -> bf16 ----------------
__global__ __launch_bounds__(256) void k_cvt_x(const float* __restrict__ x,
                                               u16* __restrict__ xb) {
  size_t g = (size_t)blockIdx.x * 256 + threadIdx.x;
  size_t base = g * 4;
  float4 f = *(const float4*)(x + base);
  u32 lo = (u32)f2bf(f.x) | ((u32)f2bf(f.y) << 16);
  u32 hi = (u32)f2bf(f.z) | ((u32)f2bf(f.w) << 16);
  uint2 o; o.x = lo; o.y = hi;
  *(uint2*)(xb + base) = o;
}

// ---------------- kernel 0b: W (K x N) -> Wt bf16 (N x K), 3 weights -----
__global__ __launch_bounds__(256) void k_cvt_w(const float* __restrict__ Wq,
                                               const float* __restrict__ Wk,
                                               const float* __restrict__ Wv,
                                               u16* __restrict__ wt) {
  size_t idx = (size_t)blockIdx.x * 256 + threadIdx.x;
  int w   = (int)(idx >> 20);
  int rem = (int)(idx & 1048575);
  int n = rem >> 10;
  int k = rem & 1023;
  const float* W = (w == 0) ? Wq : (w == 1) ? Wk : Wv;
  wt[idx] = f2bf(W[(size_t)k * DMODEL + n]);
}

// ---------------- kernel 1: QKV GEMM (async double-buffered) -------------
// Q,K stored head-major [B,H,S,DH]; V stored TRANSPOSED [B,H,DH,S] so the
// attention kernel can TDM/async its V tiles without an in-kernel transpose.
__global__ __launch_bounds__(256) void k_qkv(const u16* __restrict__ XB,
                                             const u16* __restrict__ WT,
                                             const float* __restrict__ bq,
                                             const float* __restrict__ bk,
                                             const float* __restrict__ bv,
                                             u16* __restrict__ QH,
                                             u16* __restrict__ KH,
                                             u16* __restrict__ VT) {
  __shared__ __align__(16) u16 As[2][128 * 32];
  __shared__ __align__(16) u16 Bs[2][64 * 32];

  int m0   = blockIdx.x * 128;
  int n0   = blockIdx.y * 64;
  int wsel = n0 >> 10;
  int ncol = n0 & 1023;

  const float* bias = (wsel == 0) ? bq : (wsel == 1) ? bk : bv;
  u16* OUT          = (wsel == 0) ? QH : (wsel == 1) ? KH : VT;

  int tid  = threadIdx.x;
  int wave = tid >> 5;
  int lane = tid & 31;

  const u16* wbase = WT + ((size_t)wsel << 20);

  // staging: A tile 128x32 (2 chunks/thread), B tile 64x32 (1 chunk/thread)
  int arow0 = tid >> 2,          acol0 = tid & 3;          // chunks 0..255
  int arow1 = (tid + 256) >> 2,  acol1 = tid & 3;          // chunks 256..511
  int brow  = tid >> 2,          bcol  = tid & 3;

#define STAGE_AB(buf, kk)                                                          \
  do {                                                                             \
    async_cp16(XB + (size_t)(m0 + arow0) * DMODEL + (kk) + acol0 * 8,              \
               &As[buf][arow0 * 32 + acol0 * 8]);                                  \
    async_cp16(XB + (size_t)(m0 + arow1) * DMODEL + (kk) + acol1 * 8,              \
               &As[buf][arow1 * 32 + acol1 * 8]);                                  \
    async_cp16(wbase + (size_t)(ncol + brow) * DMODEL + (kk) + bcol * 8,           \
               &Bs[buf][brow * 32 + bcol * 8]);                                    \
  } while (0)

  v8f acc[4];
  for (int i = 0; i < 4; ++i) acc[i] = v8f_zero();

  STAGE_AB(0, 0);

  for (int kk = 0; kk < DMODEL; kk += 32) {
    int buf = (kk >> 5) & 1;
    wait_async0();
    __syncthreads();
    if (kk + 32 < DMODEL) STAGE_AB(buf ^ 1, kk + 32);

    v16bf a = load_op(&As[buf][(wave * 16) * 32], 32);
#pragma unroll
    for (int nt = 0; nt < 4; ++nt) {
      v16bf b = load_op(&Bs[buf][(nt * 16) * 32], 32);
      acc[nt] = wmma_bf16(a, b, acc[nt]);
    }
  }
#undef STAGE_AB

  // epilogue: +bias, scatter
  int lm = (lane >> 4) << 3;
#pragma unroll
  for (int nt = 0; nt < 4; ++nt) {
    int n  = ncol + nt * 16 + (lane & 15);
    float bb = bias[n];
    int h = n >> 6, d = n & 63;
#pragma unroll
    for (int r = 0; r < 8; ++r) {
      int m = m0 + wave * 16 + lm + r;
      int b = m >> 11, s = m & 2047;
      size_t off;
      if (wsel == 2) off = ((size_t)(b * NH + h) * DH + d) * SEQ + s;      // V^T
      else           off = (((size_t)(b * NH + h) * SEQ + s) << 6) + d;    // Q,K
      OUT[off] = f2bf(acc[nt][r] + bb);
    }
  }
}

// ---------------- kernel 2: flash attention per (b,h) --------------------
// grid.x = SEQ/128 query tiles, grid.y = B*H. 8 waves x 16 q-rows each.
// K tiles staged with async-to-LDS; V^T tiles staged with the TDM (one
// 64x32 strided 2-D descriptor per tile), both double-buffered.
__global__ __launch_bounds__(256) void k_attn(const u16* __restrict__ QH,
                                              const u16* __restrict__ KH,
                                              const u16* __restrict__ VT,
                                              float* __restrict__ Od) {
  __shared__ __align__(16) u16 Qs[128 * 64];
  __shared__ __align__(16) u16 Ks[2][32 * 64];
  __shared__ __align__(16) u16 Vts[2][64 * 32];
  __shared__ __align__(16) u16 Ps[8][16 * 32];

  int bh = blockIdx.y;
  int q0 = blockIdx.x * 128;
  const u16* Qp  = QH + ((size_t)bh * SEQ) * DH;
  const u16* Kp  = KH + ((size_t)bh * SEQ) * DH;
  const u16* Vtp = VT + ((size_t)bh * DH) * SEQ;   // [DH][SEQ]

  int tid = threadIdx.x, wave = tid >> 5, lane = tid & 31;

  // ---- staging helpers ----
  int krow = tid >> 3, kcol = tid & 7;     // K tile: 32x64, 256 chunks
#define STAGE_K(buf, kt)                                                        \
  async_cp16(Kp + (size_t)((kt) + krow) * DH + kcol * 8,                        \
             &Ks[buf][krow * 64 + kcol * 8])

#if HAVE_TDM
#define STAGE_V(buf, kt)                                                        \
  do {                                                                          \
    if (wave == 0)                                                              \
      tdm_load_2d(lds_byte_off(&Vts[buf][0]), Vtp + (kt),                       \
                  /*tile_d0=*/32, /*tile_d1=*/64,                               \
                  /*tens_d0=*/SEQ, /*tens_d1=*/DH, /*stride0=*/SEQ);            \
  } while (0)
#define WAIT_V()  do { if (wave == 0) wait_tensor0(); } while (0)
#else
  int vrow = tid >> 2, vcol = tid & 3;     // V^T tile: 64x32, 256 chunks
#define STAGE_V(buf, kt)                                                        \
  async_cp16(Vtp + (size_t)vrow * SEQ + (kt) + vcol * 8,                        \
             &Vts[buf][vrow * 32 + vcol * 8])
#define WAIT_V()  do { } while (0)
#endif

  // stage Q tile (128x64 bf16, 4 chunks/thread) + first K/V tiles
#pragma unroll
  for (int i = 0; i < 4; ++i) {
    int c = tid + i * 256;
    int row = c >> 3, col = c & 7;
    async_cp16(Qp + (size_t)(q0 + row) * DH + col * 8, Qs + row * 64 + col * 8);
  }
  STAGE_K(0, 0);
  STAGE_V(0, 0);
  wait_async0();
  WAIT_V();
  __syncthreads();

  v16bf qa0 = load_op(Qs + wave * 16 * 64 + 0, 64);
  v16bf qa1 = load_op(Qs + wave * 16 * 64 + 32, 64);

  float mrun[8], lrun[8];
  v8f oacc[4];
#pragma unroll
  for (int r = 0; r < 8; ++r) { mrun[r] = -1e30f; lrun[r] = 0.f; }
  for (int i = 0; i < 4; ++i) oacc[i] = v8f_zero();

  const float scale = 0.03125f;  // 1/sqrt(1024) per reference

  for (int kt = 0; kt < SEQ; kt += 32) {
    int buf = (kt >> 5) & 1;
    if (kt) {
      wait_async0();
      WAIT_V();
      __syncthreads();
    }
    if (kt + 32 < SEQ) {
      STAGE_K(buf ^ 1, kt + 32);
      STAGE_V(buf ^ 1, kt + 32);
    }

    // scores: 16 q-rows x 32 keys, K-dim = 64
    v8f s0 = v8f_zero(), s1 = v8f_zero();
    s0 = wmma_bf16(qa0, load_op(&Ks[buf][0 * 16 * 64 + 0], 64), s0);
    s0 = wmma_bf16(qa1, load_op(&Ks[buf][0 * 16 * 64 + 32], 64), s0);
    s1 = wmma_bf16(qa0, load_op(&Ks[buf][1 * 16 * 64 + 0], 64), s1);
    s1 = wmma_bf16(qa1, load_op(&Ks[buf][1 * 16 * 64 + 32], 64), s1);

    // online softmax; C row m = 8*(lane>>4)+r lives across 16 lanes
#pragma unroll
    for (int r = 0; r < 8; ++r) {
      float a = s0[r] * scale;
      float b = s1[r] * scale;
      float mx = fmaxf(a, b);
      for (int off = 8; off >= 1; off >>= 1)
        mx = fmaxf(mx, __shfl_xor(mx, off, 32));
      float mnew  = fmaxf(mrun[r], mx);
      float alpha = __expf(mrun[r] - mnew);
      float pa = __expf(a - mnew);
      float pb = __expf(b - mnew);
      float ls = pa + pb;
      for (int off = 8; off >= 1; off >>= 1)
        ls += __shfl_xor(ls, off, 32);
      lrun[r] = lrun[r] * alpha + ls;
      mrun[r] = mnew;
#pragma unroll
      for (int nt = 0; nt < 4; ++nt) oacc[nt][r] *= alpha;
      int m = ((lane >> 4) << 3) + r;
      Ps[wave][m * 32 + (lane & 15)]      = f2bf(pa);
      Ps[wave][m * 32 + 16 + (lane & 15)] = f2bf(pb);
    }
    // wave-private LDS round-trip (DS ops in-order within a wave)
    v16bf pA = load_op(&Ps[wave][0], 32);
#pragma unroll
    for (int nt = 0; nt < 4; ++nt) {
      v16bf vb = load_op(&Vts[buf][nt * 16 * 32], 32);
      oacc[nt] = wmma_bf16(pA, vb, oacc[nt]);
    }
  }
#undef STAGE_K
#undef STAGE_V
#undef WAIT_V

  // finalize: divide by row sums, write fp32 attention output to d_out
  int b = bh >> 4, h = bh & 15;
#pragma unroll
  for (int nt = 0; nt < 4; ++nt) {
    int d = nt * 16 + (lane & 15);
#pragma unroll
    for (int r = 0; r < 8; ++r) {
      int m = q0 + wave * 16 + ((lane >> 4) << 3) + r;
      float o = oacc[nt][r] / lrun[r];
      Od[((size_t)(b * SEQ + m)) * DMODEL + h * DH + d] = o;
    }
  }
}

// ---------------- kernel 3: residual + LayerNorm (in-place on d_out) -----
__global__ __launch_bounds__(256) void k_ln(const float* __restrict__ x,
                                            const float* __restrict__ gamma,
                                            const float* __restrict__ beta,
                                            float* __restrict__ out) {
  __shared__ float red[256];
  int row = blockIdx.x;
  int tid = threadIdx.x;
  const float* xr = x + (size_t)row * DMODEL;
  float* orow     = out + (size_t)row * DMODEL;

  float v[4];
  float s = 0.f;
#pragma unroll
  for (int i = 0; i < 4; ++i) {
    int idx = tid + i * 256;
    v[i] = orow[idx] + xr[idx];
    s += v[i];
  }
  red[tid] = s;
  __syncthreads();
  for (int o = 128; o > 0; o >>= 1) {
    if (tid < o) red[tid] += red[tid + o];
    __syncthreads();
  }
  float mu = red[0] * (1.0f / DMODEL);
  __syncthreads();

  float vs = 0.f;
#pragma unroll
  for (int i = 0; i < 4; ++i) {
    float d = v[i] - mu;
    vs += d * d;
  }
  red[tid] = vs;
  __syncthreads();
  for (int o = 128; o > 0; o >>= 1) {
    if (tid < o) red[tid] += red[tid + o];
    __syncthreads();
  }
  float var = red[0] * (1.0f / DMODEL);
  float inv = rsqrtf(var + 1e-5f);
#pragma unroll
  for (int i = 0; i < 4; ++i) {
    int idx = tid + i * 256;
    orow[idx] = (v[i] - mu) * inv * gamma[idx] + beta[idx];
  }
}

// ---------------- launch ----------------
extern "C" void kernel_launch(void* const* d_in, const int* in_sizes, int n_in,
                              void* d_out, int out_size, void* d_ws, size_t ws_size,
                              hipStream_t stream) {
  (void)in_sizes; (void)n_in; (void)out_size; (void)ws_size;

  const float* x  = (const float*)d_in[0];
  const float* Wq = (const float*)d_in[1];
  const float* bq = (const float*)d_in[2];
  const float* Wk = (const float*)d_in[3];
  const float* bk = (const float*)d_in[4];
  const float* Wv = (const float*)d_in[5];
  const float* bv = (const float*)d_in[6];
  const float* ga = (const float*)d_in[7];
  const float* be = (const float*)d_in[8];
  float* out = (float*)d_out;

  // workspace layout (bytes)
  char* ws = (char*)d_ws;
  u16* XB = (u16*)(ws);                               // 8192*1024   bf16 = 16 MB
  u16* WT = (u16*)(ws + 16777216);                    // 3*1024*1024 bf16 =  6 MB
  u16* QH = (u16*)(ws + 23068672);                    // [B,H,S,DH]  bf16 = 16 MB
  u16* KH = (u16*)(ws + 39845888);                    // [B,H,S,DH]
  u16* VT = (u16*)(ws + 56623104);                    // [B,H,DH,S]  (transposed)

  k_cvt_x<<<MTOT * DMODEL / (256 * 4), 256, 0, stream>>>(x, XB);
  k_cvt_w<<<3 * DMODEL * DMODEL / 256, 256, 0, stream>>>(Wq, Wk, Wv, WT);
  k_qkv<<<dim3(MTOT / 128, 3 * DMODEL / 64), 256, 0, stream>>>(
      XB, WT, bq, bk, bv, QH, KH, VT);
  k_attn<<<dim3(SEQ / 128, BATCH * NH), 256, 0, stream>>>(QH, KH, VT, out);
  k_ln<<<MTOT, 256, 0, stream>>>(x, ga, be, out);
}
